// HGTModel_29618094473872
// MI455X (gfx1250) — matrix operand order repaired and checked by reading, not compiled, not run on hardware
//
#include <hip/hip_runtime.h>
#include <hip/hip_bf16.h>
#include <cstddef>
#include <cstdint>

// ---------------------------------------------------------------------------
// HGT (heterogeneous graph transformer) forward on gfx1250.
// Dense 64x64 linears -> v_wmma_f32_16x16x32_f16 (f16 in, f32 accumulate),
// with LDS-staged f16 tiles (ds_load_b128 fragments) and fused QKV so node
// features are read once per layer. Edge softmax/aggregation -> atomic
// segment ops (bandwidth bound).
// ---------------------------------------------------------------------------

typedef _Float16 v16h __attribute__((ext_vector_type(16)));
typedef float    v8f  __attribute__((ext_vector_type(8)));

#define HDIM 64
#define HEADS 8
#define DD 8

// ---- order-preserving float <-> u32 encoding for atomic max -----------------
__device__ __forceinline__ unsigned fenc(float f) {
    unsigned u = __float_as_uint(f);
    return (u & 0x80000000u) ? ~u : (u | 0x80000000u);
}
__device__ __forceinline__ float fdec(unsigned e) {
    unsigned u = (e & 0x80000000u) ? (e & 0x7FFFFFFFu) : ~e;
    return __uint_as_float(u);
}

// ---- fills ------------------------------------------------------------------
__global__ void fill_f_kernel(float* p, float v, size_t n) {
    size_t t = (size_t)blockIdx.x * blockDim.x + threadIdx.x;
    if (t < n) p[t] = v;
}
__global__ void fill_u_kernel(unsigned* p, unsigned v, size_t n) {
    size_t t = (size_t)blockIdx.x * blockDim.x + threadIdx.x;
    if (t < n) p[t] = v;
}

// ---- node encoder: h = relu(x @ W + b), tiny fan-in (1/2/7) -> plain VALU ---
__global__ void encode_kernel(const float* __restrict__ x, const float* __restrict__ W,
                              const float* __restrict__ b, float* __restrict__ h,
                              int N, int in_dim) {
    int t = blockIdx.x * blockDim.x + threadIdx.x;
    if (t >= N * HDIM) return;
    int n = t >> 6, c = t & 63;
    float acc = b[c];
    for (int i = 0; i < in_dim; ++i)
        acc = fmaf(x[(size_t)n * in_dim + i], W[i * HDIM + c], acc);
    h[t] = fmaxf(acc, 0.0f);
}

// ---- edge-attr projection + scatter-add into src & dst node features --------
__global__ void edge_proj_kernel(const int* __restrict__ src, const int* __restrict__ dst,
                                 const float* __restrict__ eattr,
                                 const float* __restrict__ w, const float* __restrict__ b,
                                 float* __restrict__ hsrc, float* __restrict__ hdst, int E) {
    int t = blockIdx.x * blockDim.x + threadIdx.x;
    if (t >= E * HDIM) return;
    int e = t >> 6, c = t & 63;
    float proj = fmaf(eattr[e], w[c], b[c]);
    atomicAdd(hsrc + (size_t)src[e] * HDIM + c, proj);
    atomicAdd(hdst + (size_t)dst[e] * HDIM + c, proj);
}

// ---- fused multi-output WMMA GEMM ------------------------------------------
// out_o[N,64] = act(beta*(act_in(in) @ W_o + b_o) + (1-beta)*skip), o < NOUT.
// Block = 128 threads (4 wave32); block owns 16 rows x all 64 cols; each wave
// owns one 16-col stripe. A tile (16x64) and W_o (64x64, transposed) are
// staged once in LDS as f16 -> fragments are two ds_load_b128 each, and the
// f32->f16 conversion happens once per element instead of once per wave.
// N % 16 == 0 for every call site, so EXEC is all-ones at each WMMA.
template <int NOUT, int GELU_IN, int RELU>
__global__ __launch_bounds__(128)
void gemm64_fused_kernel(const float* __restrict__ in,
                         const float* __restrict__ W0, const float* __restrict__ W1,
                         const float* __restrict__ W2,
                         const float* __restrict__ b0, const float* __restrict__ b1,
                         const float* __restrict__ b2,
                         float* __restrict__ out0, float* __restrict__ out1,
                         float* __restrict__ out2, int N,
                         const float* __restrict__ skip_src,
                         const float* __restrict__ skip_gate) {
    __shared__ __align__(16) _Float16 As[16 * HDIM];          // A tile, row-major
    __shared__ __align__(16) _Float16 Ws[NOUT][HDIM * HDIM];  // W^T: Ws[o][col*64+k]
    const float* Wg[3] = {W0, W1, W2};
    const float* bg[3] = {b0, b1, b2};
    float*       og[3] = {out0, out1, out2};

    const int tid  = threadIdx.x;
    const int row0 = blockIdx.x * 16;

    // stage A tile (coalesced), optional exact-gelu on the fly
    for (int i = tid; i < 16 * HDIM; i += 128) {
        float x = in[(size_t)row0 * HDIM + i];
        if (GELU_IN) x = 0.5f * x * (1.0f + erff(x * 0.70710678118654752f));
        As[i] = (_Float16)x;
    }
    // stage W transposed (coalesced global reads)
#pragma unroll
    for (int o = 0; o < NOUT; ++o)
        for (int i = tid; i < HDIM * HDIM; i += 128) {
            int r = i >> 6, c = i & 63;
            Ws[o][c * HDIM + r] = (_Float16)Wg[o][i];
        }
    __syncthreads();

    const int wave = tid >> 5;
    const int lane = tid & 31;
    const int half = lane >> 4;
    const int m    = lane & 15;
    const int col  = wave * 16 + m;

    v8f acc[NOUT];
#pragma unroll
    for (int o = 0; o < NOUT; ++o) acc[o] = {};

#pragma unroll
    for (int k = 0; k < HDIM; k += 32) {
        // A 16x32 f16 layout (ISA 7.12.2): elems 0..7 -> K=k+8*half+e,
        //                                  elems 8..15 -> K=k+16+8*half+e
        v16h a;
        const _Float16* ap = &As[m * HDIM + k + 8 * half];
#pragma unroll
        for (int e = 0; e < 8; ++e) { a[e] = ap[e]; a[e + 8] = ap[16 + e]; }
#pragma unroll
        for (int o = 0; o < NOUT; ++o) {
            // B 32x16 f16 layout: elems 0..7 -> K=k+8*half+e, elems 8..15 -> K=k+16+8*half+e, N = lane&15
            v16h b;
            const _Float16* bp = &Ws[o][col * HDIM + k + 8 * half];
#pragma unroll
            for (int e = 0; e < 8; ++e) { b[e] = bp[e]; b[e + 8] = bp[16 + e]; }
            acc[o] = __builtin_amdgcn_wmma_f32_16x16x32_f16(
                /*neg_a=*/false, a, /*neg_b=*/false, b,
                /*c_mod=*/(short)0, acc[o], /*reuse_a=*/false, /*reuse_b=*/false);
        }
    }

    float beta = 1.0f, invb = 0.0f;
    if (skip_gate) {
        float g = skip_gate[0];
        beta = 1.0f / (1.0f + __expf(-g));
        invb = 1.0f - beta;
    }
    // C/D layout: VGPR i -> row = i + 8*half, col = lane&15
#pragma unroll
    for (int o = 0; o < NOUT; ++o) {
#pragma unroll
        for (int i = 0; i < 8; ++i) {
            int r = row0 + i + 8 * half;
            float v = acc[o][i] + bg[o][col];
            if (skip_src && o == 0) v = beta * v + invb * skip_src[(size_t)r * HDIM + col];
            if (RELU) v = fmaxf(v, 0.0f);
            og[o][(size_t)r * HDIM + col] = v;
        }
    }
}

// ---- per-(edge,head) attention logit + segment max --------------------------
__global__ __launch_bounds__(256)
void edge_logit_kernel(const int* __restrict__ src, const int* __restrict__ dst, int E,
                       const float* __restrict__ Kf, const float* __restrict__ Qf,
                       const float* __restrict__ Arel, const float* __restrict__ prel,
                       float* __restrict__ logits, unsigned* __restrict__ mx) {
    __shared__ float A[HEADS * DD * DD];
    for (int i = threadIdx.x; i < HEADS * DD * DD; i += blockDim.x) A[i] = Arel[i];
    __syncthreads();
    int t = blockIdx.x * blockDim.x + threadIdx.x;
    if (t >= E * HEADS) return;
    int e = t >> 3, h = t & 7;
    int d_node = dst[e];
    const float* kv = Kf + (size_t)src[e] * HDIM + h * DD;
    const float* qv = Qf + (size_t)d_node * HDIM + h * DD;
    float kd[DD];
#pragma unroll
    for (int d = 0; d < DD; ++d) kd[d] = kv[d];
    float logit = 0.0f;
#pragma unroll
    for (int f = 0; f < DD; ++f) {
        float kt = 0.0f;
#pragma unroll
        for (int d = 0; d < DD; ++d) kt = fmaf(kd[d], A[h * 64 + d * DD + f], kt);
        logit = fmaf(qv[f], kt, logit);
    }
    logit *= prel[h] * 0.35355339059327373f;  // 1/sqrt(D)
    logits[t] = logit;
    atomicMax(mx + (size_t)d_node * HEADS + h, fenc(logit));
}

// ---- exp(logit - max) + segment sum -----------------------------------------
__global__ __launch_bounds__(256)
void edge_exp_kernel(const int* __restrict__ dst, int E,
                     float* __restrict__ logits, const unsigned* __restrict__ mx,
                     float* __restrict__ den) {
    int t = blockIdx.x * blockDim.x + threadIdx.x;
    if (t >= E * HEADS) return;
    int e = t >> 3, h = t & 7;
    int d_node = dst[e];
    float m = fdec(mx[(size_t)d_node * HEADS + h]);
    float ex = __expf(logits[t] - m);
    logits[t] = ex;
    atomicAdd(den + (size_t)d_node * HEADS + h, ex);
}

// ---- weighted value aggregate (recomputes vt = vd @ m_rel) ------------------
__global__ __launch_bounds__(256)
void edge_agg_kernel(const int* __restrict__ src, const int* __restrict__ dst, int E,
                     const float* __restrict__ Vf, const float* __restrict__ Mrel,
                     const float* __restrict__ logits, const float* __restrict__ den,
                     float* __restrict__ agg) {
    __shared__ float M[HEADS * DD * DD];
    for (int i = threadIdx.x; i < HEADS * DD * DD; i += blockDim.x) M[i] = Mrel[i];
    __syncthreads();
    int t = blockIdx.x * blockDim.x + threadIdx.x;
    if (t >= E * HEADS) return;
    int e = t >> 3, h = t & 7;
    int d_node = dst[e];
    float w = logits[t] / fmaxf(den[(size_t)d_node * HEADS + h], 1e-16f);
    const float* vv = Vf + (size_t)src[e] * HDIM + h * DD;
    float vd[DD];
#pragma unroll
    for (int d = 0; d < DD; ++d) vd[d] = vv[d];
#pragma unroll
    for (int f = 0; f < DD; ++f) {
        float vt = 0.0f;
#pragma unroll
        for (int d = 0; d < DD; ++d) vt = fmaf(vd[d], M[h * 64 + d * DD + f], vt);
        atomicAdd(agg + (size_t)d_node * HDIM + h * DD + f, vt * w);
    }
}

// ---- final head2: [N,64] @ [64,3] + b ---------------------------------------
__global__ void head2_kernel(const float* __restrict__ z, const float* __restrict__ W,
                             const float* __restrict__ b, float* __restrict__ out, int N) {
    int n = blockIdx.x * blockDim.x + threadIdx.x;
    if (n >= N) return;
    float acc0 = b[0], acc1 = b[1], acc2 = b[2];
    const float* zr = z + (size_t)n * HDIM;
#pragma unroll
    for (int i = 0; i < HDIM; ++i) {
        float zi = zr[i];
        acc0 = fmaf(zi, W[i * 3 + 0], acc0);
        acc1 = fmaf(zi, W[i * 3 + 1], acc1);
        acc2 = fmaf(zi, W[i * 3 + 2], acc2);
    }
    out[(size_t)n * 3 + 0] = acc0;
    out[(size_t)n * 3 + 1] = acc1;
    out[(size_t)n * 3 + 2] = acc2;
}

// ---------------------------------------------------------------------------
// Host orchestration
// ---------------------------------------------------------------------------
static inline unsigned gsz(size_t n, int b) { return (unsigned)((n + b - 1) / b); }

extern "C" void kernel_launch(void* const* d_in, const int* in_sizes, int n_in,
                              void* d_out, int out_size, void* d_ws, size_t ws_size,
                              hipStream_t stream) {
    (void)in_sizes; (void)n_in; (void)out_size;

    // node types: 0=service, 1=queue, 2=topic  (sorted-key param order: q,s,t)
    const size_t Nn[3]       = {200000, 100000, 50000};
    const int    sortedNt[3] = {1, 0, 2};  // nt -> index in sorted {queue,service,topic}

    struct Rel { int src_i, dst_i, E, stype, ttype; };
    const Rel rels[5] = {
        {3, 4, 400000, 0, 0},    // calls:      service -> service
        {6, 7, 200000, 0, 1},    // produces:   service -> queue
        {9, 10, 150000, 1, 2},   // publishes:  queue   -> topic
        {12, 13, 150000, 2, 0},  // subscribes: topic   -> service
        {15, 16, 100000, 1, 0},  // consumes:   queue   -> service
    };
    // sorted-key rel order: calls, consumes, produces, publishes, subscribes
    const int relSorted[5] = {0, 2, 3, 4, 1};

    // ---- params pytree leaf indices (sorted-key flatten), base 18 ----
    // edge_enc: 18..27 (b,w per sorted rel); enc: 28..33; head1: 34,35; head2: 36,37
    // hgt1: 38..79, hgt2: 80..121 with sublayout below.
    auto F = [&](int i) { return (const float*)d_in[i]; };
    auto I = [&](int i) { return (const int*)d_in[i]; };

    const int encB = 28, h1B = 34, h2B = 36;
    auto hgtBase  = [&](int layer) { return layer == 0 ? 38 : 80; };
    auto idxArel  = [&](int L, int r)  { return hgtBase(L) + 0  + relSorted[r]; };
    auto idxKb    = [&](int L, int nt) { return hgtBase(L) + 5  + 2 * sortedNt[nt]; };
    auto idxKw    = [&](int L, int nt) { return idxKb(L, nt) + 1; };
    auto idxMrel  = [&](int L, int r)  { return hgtBase(L) + 11 + relSorted[r]; };
    auto idxOutb  = [&](int L, int nt) { return hgtBase(L) + 16 + 2 * sortedNt[nt]; };
    auto idxOutw  = [&](int L, int nt) { return idxOutb(L, nt) + 1; };
    auto idxPrel  = [&](int L, int r)  { return hgtBase(L) + 22 + relSorted[r]; };
    auto idxQb    = [&](int L, int nt) { return hgtBase(L) + 27 + 2 * sortedNt[nt]; };
    auto idxQw    = [&](int L, int nt) { return idxQb(L, nt) + 1; };
    auto idxSkip  = [&](int L, int nt) { return hgtBase(L) + 33 + sortedNt[nt]; };
    auto idxVb    = [&](int L, int nt) { return hgtBase(L) + 36 + 2 * sortedNt[nt]; };
    auto idxVw    = [&](int L, int nt) { return idxVb(L, nt) + 1; };

    // ---- workspace carve-up (floats) ----
    float* base = (float*)d_ws;
    size_t off = 0;
    auto alloc = [&](size_t nf) { float* p = base + off; off += nf; return p; };

    float *h[3], *hn[3], *Kf[3], *Qf[3], *Vf[3], *agg[3], *den[3];
    unsigned* mx[3];
    for (int nt = 0; nt < 3; ++nt) h[nt]   = alloc(Nn[nt] * HDIM);
    for (int nt = 0; nt < 3; ++nt) hn[nt]  = alloc(Nn[nt] * HDIM);
    for (int nt = 0; nt < 3; ++nt) Kf[nt]  = alloc(Nn[nt] * HDIM);
    for (int nt = 0; nt < 3; ++nt) Qf[nt]  = alloc(Nn[nt] * HDIM);
    for (int nt = 0; nt < 3; ++nt) Vf[nt]  = alloc(Nn[nt] * HDIM);
    for (int nt = 0; nt < 3; ++nt) agg[nt] = alloc(Nn[nt] * HDIM);
    for (int nt = 0; nt < 3; ++nt) den[nt] = alloc(Nn[nt] * HEADS);
    for (int nt = 0; nt < 3; ++nt) mx[nt]  = (unsigned*)alloc(Nn[nt] * HEADS);
    float* logits[5];
    for (int r = 0; r < 5; ++r) logits[r] = alloc((size_t)rels[r].E * HEADS);
    float* z = alloc(Nn[0] * HDIM);
    if (off * sizeof(float) > ws_size) return;  // deterministic guard

    const int B = 256;

    // ---- 1. node encoders: h = relu(x @ enc_w + enc_b) ----
    const int in_dims[3] = {7, 2, 1};
    for (int nt = 0; nt < 3; ++nt) {
        size_t n = Nn[nt] * HDIM;
        encode_kernel<<<gsz(n, B), B, 0, stream>>>(
            F(nt), F(encB + 1 + 2 * sortedNt[nt]), F(encB + 2 * sortedNt[nt]),
            h[nt], (int)Nn[nt], in_dims[nt]);
    }

    // ---- 2. edge-attr scatter-add into src & dst features ----
    for (int r = 0; r < 5; ++r) {
        const Rel& R = rels[r];
        size_t n = (size_t)R.E * HDIM;
        edge_proj_kernel<<<gsz(n, B), B, 0, stream>>>(
            I(R.src_i), I(R.dst_i), F(R.src_i + 2),
            F(19 + 2 * relSorted[r]), F(18 + 2 * relSorted[r]),
            h[R.stype], h[R.ttype], R.E);
    }

    // ---- 3. two HGT conv layers ----
    for (int L = 0; L < 2; ++L) {
        // fused K/Q/V projection: read h once, 6 WMMA per wave k-loop
        for (int nt = 0; nt < 3; ++nt) {
            gemm64_fused_kernel<3, 0, 0><<<(unsigned)(Nn[nt] / 16), 128, 0, stream>>>(
                h[nt],
                F(idxKw(L, nt)), F(idxQw(L, nt)), F(idxVw(L, nt)),
                F(idxKb(L, nt)), F(idxQb(L, nt)), F(idxVb(L, nt)),
                Kf[nt], Qf[nt], Vf[nt], (int)Nn[nt], nullptr, nullptr);
        }
        // reset segment state
        for (int nt = 0; nt < 3; ++nt) {
            fill_u_kernel<<<gsz(Nn[nt] * HEADS, B), B, 0, stream>>>(mx[nt], 0u, Nn[nt] * HEADS);
            fill_f_kernel<<<gsz(Nn[nt] * HEADS, B), B, 0, stream>>>(den[nt], 0.0f, Nn[nt] * HEADS);
            fill_f_kernel<<<gsz(Nn[nt] * HDIM, B), B, 0, stream>>>(agg[nt], 0.0f, Nn[nt] * HDIM);
        }
        // pass 1: logits + segment max (across ALL relations into dst type)
        for (int r = 0; r < 5; ++r) {
            const Rel& R = rels[r];
            size_t n = (size_t)R.E * HEADS;
            edge_logit_kernel<<<gsz(n, B), B, 0, stream>>>(
                I(R.src_i), I(R.dst_i), R.E, Kf[R.stype], Qf[R.ttype],
                F(idxArel(L, r)), F(idxPrel(L, r)), logits[r], mx[R.ttype]);
        }
        // pass 2: exp + segment sum
        for (int r = 0; r < 5; ++r) {
            const Rel& R = rels[r];
            size_t n = (size_t)R.E * HEADS;
            edge_exp_kernel<<<gsz(n, B), B, 0, stream>>>(
                I(R.dst_i), R.E, logits[r], mx[R.ttype], den[R.ttype]);
        }
        // pass 3: weighted aggregate of m_rel-transformed values
        for (int r = 0; r < 5; ++r) {
            const Rel& R = rels[r];
            size_t n = (size_t)R.E * HEADS;
            edge_agg_kernel<<<gsz(n, B), B, 0, stream>>>(
                I(R.src_i), I(R.dst_i), R.E, Vf[R.stype], F(idxMrel(L, r)),
                logits[r], den[R.ttype], agg[R.ttype]);
        }
        // out-linear with gelu fused into A-tile staging, gated skip, relu
        for (int nt = 0; nt < 3; ++nt) {
            gemm64_fused_kernel<1, 1, 1><<<(unsigned)(Nn[nt] / 16), 128, 0, stream>>>(
                agg[nt], F(idxOutw(L, nt)), nullptr, nullptr,
                F(idxOutb(L, nt)), nullptr, nullptr,
                hn[nt], nullptr, nullptr, (int)Nn[nt],
                h[nt], F(idxSkip(L, nt)));
        }
        for (int nt = 0; nt < 3; ++nt) { float* t = h[nt]; h[nt] = hn[nt]; hn[nt] = t; }
    }

    // ---- 4. heads: z = relu(h_service @ W1 + b1); out = z @ W2 + b2 ----
    gemm64_fused_kernel<1, 0, 1><<<(unsigned)(Nn[0] / 16), 128, 0, stream>>>(
        h[0], F(h1B + 1), nullptr, nullptr, F(h1B), nullptr, nullptr,
        z, nullptr, nullptr, (int)Nn[0], nullptr, nullptr);
    head2_kernel<<<gsz(Nn[0], B), B, 0, stream>>>(z, F(h2B + 1), F(h2B),
                                                  (float*)d_out, (int)Nn[0]);
}
